// Model2HR_31731218383004
// MI455X (gfx1250) — matrix operand order repaired and checked by reading, compiled, not run on hardware
//
#include <hip/hip_runtime.h>
#include <hip/hip_bf16.h>

typedef __attribute__((ext_vector_type(16))) _Float16 v16h;
typedef __attribute__((ext_vector_type(8)))  float    v8f;
typedef __attribute__((ext_vector_type(4)))  unsigned int uint4v;

#define RT 8            // B*T rows
#define MT 4            // M-tiles (of 16) per block
#define APAD 40         // padded row stride (halfs): 80B, 16B-aligned, conflict-free

// ---------------------------------------------------------------------------
// Fused gather (7-ring) + GEMM + bias + ReLU using v_wmma_f32_16x16x32_f16,
// software-pipelined with double-buffered LDS tiles and fast/guarded staging.
// X: [RT, N, Cin] f32, neigh: [N,7] i32, W: [7*Cin, Cout] f32, b: [Cout] f32
// Y: [RT, N, Cout] f32.   grid.x = ceil(RT*N/64), grid.y = Cout/64, block=128.
// ---------------------------------------------------------------------------
__global__ __launch_bounds__(128) void sconv16(
    const float* __restrict__ X, const int* __restrict__ neigh,
    const float* __restrict__ W, const float* __restrict__ bias,
    float* __restrict__ Y, int N, int Cin, int Cout, int relu)
{
  const int M    = RT * N;
  const int Ktot = 7 * Cin;
  const int wave = threadIdx.x >> 5;
  const int lane = threadIdx.x & 31;
  const int m0   = blockIdx.x * (16 * MT);
  const int co0  = blockIdx.y * 64 + wave * 16;
  const bool fullM = (m0 + 16 * MT) <= M;        // uniform: no row guards needed

  __shared__ long     rowBase[MT * 16][8];       // K-invariant gather bases
  __shared__ _Float16 As[2][MT][16 * APAD];      // double-buffered A staging
  __shared__ _Float16 Bs[2][4][16 * APAD];       // double-buffered B staging (per wave)

  // ---- precompute gather base offsets once per block (448 entries) ----
  for (int t = threadIdx.x; t < MT * 16 * 7; t += 128) {
    const int row = t / 7, k7 = t - row * 7;
    const int m = m0 + row;
    long bse = -1;
    if (m < M) {
      const int r = m / N, n = m - r * N;
      bse = ((long)r * N + neigh[n * 7 + k7]) * Cin;
    }
    rowBase[row][k7] = bse;
  }
  __syncthreads();

  // A staging mapping: 2 threads per row, each owns 16 contiguous K
  const int srow   = threadIdx.x >> 1;           // 0..63
  const int kkbase = (threadIdx.x & 1) * 16;     // 0 or 16
  const int stile  = srow >> 4;
  const int rloc   = srow & 15;
  // B staging mapping: col = lane&15, each lane owns 16 contiguous kk
  const int bcol   = lane & 15;
  const int bhalf  = lane >> 4;                  // kk base = bhalf*16

  float ta[16], tb[16];

  // fast path: full M-tile, full K-chunk -> unconditional loads
  auto stage_loads = [&](int kc) {
    const bool fullChunk = (kc + 32) <= Ktot;    // uniform
    {
      const int kg = kc + kkbase;
      int k7 = kg / Cin;                         // one divide per chunk
      int c  = kg - k7 * Cin;
      if (fullM && fullChunk) {
#pragma unroll
        for (int j = 0; j < 16; ++j) {
          ta[j] = X[rowBase[srow][k7] + c];
          if (++c == Cin) { c = 0; ++k7; }
        }
      } else {
#pragma unroll
        for (int j = 0; j < 16; ++j) {
          float v = 0.f;
          if (kg + j < Ktot) {
            const long bse = rowBase[srow][k7];
            if (bse >= 0) v = X[bse + c];
          }
          ta[j] = v;
          if (++c == Cin) { c = 0; ++k7; }
        }
      }
    }
    if (fullChunk) {
      const float* wp = W + (long)(kc + bhalf * 16) * Cout + co0 + bcol;
#pragma unroll
      for (int j = 0; j < 16; ++j) tb[j] = wp[(long)j * Cout];
    } else {
#pragma unroll
      for (int j = 0; j < 16; ++j) {
        const int kg = kc + bhalf * 16 + j;
        tb[j] = (kg < Ktot) ? W[(long)kg * Cout + co0 + bcol] : 0.f;
      }
    }
  };

  auto stage_store = [&](int buf) {
    unsigned ua[8], ub[8];
#pragma unroll
    for (int p = 0; p < 8; ++p) {
      union { _Float16 h[2]; unsigned u; } ca, cb;
      ca.h[0] = (_Float16)ta[2 * p]; ca.h[1] = (_Float16)ta[2 * p + 1];
      cb.h[0] = (_Float16)tb[2 * p]; cb.h[1] = (_Float16)tb[2 * p + 1];
      ua[p] = ca.u; ub[p] = cb.u;
    }
    uint4v* da = (uint4v*)&As[buf][stile][rloc * APAD + kkbase];       // 16B aligned
    da[0] = (uint4v){ua[0], ua[1], ua[2], ua[3]};
    da[1] = (uint4v){ua[4], ua[5], ua[6], ua[7]};
    uint4v* db = (uint4v*)&Bs[buf][wave][bcol * APAD + bhalf * 16];    // 16B aligned
    db[0] = (uint4v){ub[0], ub[1], ub[2], ub[3]};
    db[1] = (uint4v){ub[4], ub[5], ub[6], ub[7]};
  };

  v8f acc[MT];
#pragma unroll
  for (int u = 0; u < MT; ++u) acc[u] = (v8f){0.f,0.f,0.f,0.f,0.f,0.f,0.f,0.f};

  // ---- prologue: stage chunk 0 into buffer 0 ----
  stage_loads(0); stage_store(0);
  __syncthreads();

  int buf = 0;
  for (int k0 = 0; k0 < Ktot; k0 += 32) {
    const int kn = k0 + 32;
    const bool more = kn < Ktot;

    // 1) issue next-chunk global loads (latency hidden behind WMMAs below)
    if (more) {
      if (lane == 0 && kn + 32 < Ktot)
        __builtin_prefetch(W + (long)(kn + 32) * Cout + co0, 0, 1);
      stage_loads(kn);
    }

    // 2) compute current chunk: load all fragments first, then 4 WMMAs
    {
      const int colB = lane & 15;
      const int boff = (lane < 16) ? 0 : 16;
      v16h b = {};
#pragma unroll
      for (int j = 0; j < 16; ++j) b[j] = Bs[buf][wave][colB * APAD + boff + j];

      const int rowA = lane & 15;
      const int aoff = (lane < 16) ? 0 : 16;
      v16h afr[MT];
#pragma unroll
      for (int u = 0; u < MT; ++u) {
        v16h a = {};
#pragma unroll
        for (int j = 0; j < 8; ++j) a[j]     = As[buf][u][rowA * APAD + aoff + j];
#pragma unroll
        for (int j = 0; j < 8; ++j) a[8 + j] = As[buf][u][rowA * APAD + 8 + aoff + j];
        afr[u] = a;
      }
#pragma unroll
      for (int u = 0; u < MT; ++u)
        acc[u] = __builtin_amdgcn_wmma_f32_16x16x32_f16(
            false, afr[u], false, b, (short)0, acc[u], false, false);
    }

    // 3) convert + packed b128 store of next chunk into the other buffer
    if (more) stage_store(buf ^ 1);
    __syncthreads();
    buf ^= 1;
  }

  // ---- epilogue: bias + ReLU, C/D layout: col = lane%16, rows v / v+8 ----
  const int col   = lane & 15;
  const int rbase = (lane < 16) ? 0 : 8;
  const int co    = co0 + col;
  const float bv  = bias[co];
#pragma unroll
  for (int u = 0; u < MT; ++u) {
#pragma unroll
    for (int v = 0; v < 8; ++v) {
      const int m = m0 + u * 16 + rbase + v;
      if (m < M) {
        float y = acc[u][v] + bv;
        if (relu) y = fmaxf(y, 0.f);
        Y[(long)m * Cout + co] = y;
      }
    }
  }
}

// ---------------------------------------------------------------------------
// Ring-max pooling: Y[r,n,c] = max_{k<7} X[r, neigh[n,k], c], n < nc.
// ---------------------------------------------------------------------------
__global__ void pool_ring(const float* __restrict__ X, const int* __restrict__ neigh,
                          float* __restrict__ Y, int N, int nc, int C)
{
  const long total = (long)RT * nc * C;
  for (long t = (long)blockIdx.x * blockDim.x + threadIdx.x; t < total;
       t += (long)gridDim.x * blockDim.x) {
    const int c  = (int)(t % C);
    const long rn = t / C;
    const int n  = (int)(rn % nc);
    const int r  = (int)(rn / nc);
    float v = -3.402823466e38f;
#pragma unroll
    for (int k = 0; k < 7; ++k)
      v = fmaxf(v, X[((long)r * N + neigh[n * 7 + k]) * C + c]);
    Y[t] = v;
  }
}

// ---------------------------------------------------------------------------
// Unpool (copy coarse, max of 2 parents for new) fused with skip concat.
// ---------------------------------------------------------------------------
__global__ void unpool_concat(const float* __restrict__ U, const int* __restrict__ up,
                              const float* __restrict__ S, float* __restrict__ O,
                              int nc, int nf, int Cu, int Cs)
{
  const int  Ct = Cu + Cs;
  const long total = (long)RT * nf * Ct;
  for (long t = (long)blockIdx.x * blockDim.x + threadIdx.x; t < total;
       t += (long)gridDim.x * blockDim.x) {
    const int c  = (int)(t % Ct);
    const long rn = t / Ct;
    const int n  = (int)(rn % nf);
    const int r  = (int)(rn / nf);
    float v;
    if (c < Cu) {
      if (n < nc) {
        v = U[((long)r * nc + n) * Cu + c];
      } else {
        const int e  = n - nc;
        const int p0 = up[e * 2], p1 = up[e * 2 + 1];
        v = fmaxf(U[((long)r * nc + p0) * Cu + c],
                  U[((long)r * nc + p1) * Cu + c]);
      }
    } else {
      v = S[((long)r * nf + n) * Cs + (c - Cu)];
    }
    O[t] = v;
  }
}

// ---------------------------------------------------------------------------
// Final conv: Cout=1, Cin=64, no ReLU -> logits [RT,N].
// ---------------------------------------------------------------------------
__global__ void conv_out1(const float* __restrict__ X, const int* __restrict__ neigh,
                          const float* __restrict__ W, const float* __restrict__ b,
                          float* __restrict__ L, int N)
{
  const int t = blockIdx.x * blockDim.x + threadIdx.x;
  if (t >= RT * N) return;
  const int n = t % N, r = t / N;
  float s = b[0];
#pragma unroll
  for (int k = 0; k < 7; ++k) {
    const float* xp = X + ((long)r * N + neigh[n * 7 + k]) * 64;
    const float* wp = W + k * 64;
#pragma unroll 8
    for (int c = 0; c < 64; ++c) s += xp[c] * wp[c];
  }
  L[t] = s;
}

// ---------------------------------------------------------------------------
// Row softmax over N; one block per (b,t) row.
// ---------------------------------------------------------------------------
__global__ void softmax_rows(const float* __restrict__ L, float* __restrict__ O, int N)
{
  const int r = blockIdx.x;
  const int tid = threadIdx.x;
  const long base = (long)r * N;
  __shared__ float red[256];

  float mx = -3.402823466e38f;
  for (int i = tid; i < N; i += 256) mx = fmaxf(mx, L[base + i]);
  red[tid] = mx; __syncthreads();
  for (int s = 128; s > 0; s >>= 1) {
    if (tid < s) red[tid] = fmaxf(red[tid], red[tid + s]);
    __syncthreads();
  }
  mx = red[0]; __syncthreads();

  float sum = 0.f;
  for (int i = tid; i < N; i += 256) sum += expf(L[base + i] - mx);
  red[tid] = sum; __syncthreads();
  for (int s = 128; s > 0; s >>= 1) {
    if (tid < s) red[tid] += red[tid + s];
    __syncthreads();
  }
  const float inv = 1.f / red[0];

  for (int i = tid; i < N; i += 256) O[base + i] = expf(L[base + i] - mx) * inv;
}

// ---------------------------------------------------------------------------
// Host orchestration
// ---------------------------------------------------------------------------
static inline void launch_sconv(const float* X, const int* nb, const float* W,
                                const float* b, float* Y, int N, int Cin, int Cout,
                                int relu, hipStream_t s)
{
  dim3 g((RT * N + 16 * MT - 1) / (16 * MT), Cout / 64);
  sconv16<<<g, 128, 0, s>>>(X, nb, W, b, Y, N, Cin, Cout, relu);
}

static inline int gsz(long total) {
  long b = (total + 255) / 256;
  return (int)(b > 65535 ? 65535 : b);
}

extern "C" void kernel_launch(void* const* d_in, const int* in_sizes, int n_in,
                              void* d_out, int out_size, void* d_ws, size_t ws_size,
                              hipStream_t stream)
{
  const int N5 = 10242, N4 = 2562, N3 = 642, N2 = 162, N1 = 42;

  const float* frame  = (const float*)d_in[0];
  const int* neigh5   = (const int*)d_in[3];
  const int* neigh4   = (const int*)d_in[4];
  const int* neigh3   = (const int*)d_in[5];
  const int* neigh2   = (const int*)d_in[6];
  const int* neigh1   = (const int*)d_in[7];
  const int* up2      = (const int*)d_in[8];
  const int* up3      = (const int*)d_in[9];
  const int* up4      = (const int*)d_in[10];
  const int* up5      = (const int*)d_in[11];
  const float* W[10]; const float* bb[10];
  for (int i = 0; i < 10; ++i) {
    W[i]  = (const float*)d_in[12 + 2 * i];
    bb[i] = (const float*)d_in[13 + 2 * i];
  }

  float* ws = (float*)d_ws;
  size_t off = 0;
  auto alloc = [&](long n) { float* p = ws + off; off += (size_t)n; return p; };

  float* c0   = alloc((long)RT * N5 * 64);
  float* p0   = alloc((long)RT * N4 * 64);
  float* c1   = alloc((long)RT * N4 * 128);
  float* p1   = alloc((long)RT * N3 * 128);
  float* c2   = alloc((long)RT * N3 * 256);
  float* p2   = alloc((long)RT * N2 * 256);
  float* c3   = alloc((long)RT * N2 * 512);
  float* p3   = alloc((long)RT * N1 * 512);
  float* c4   = alloc((long)RT * N1 * 1024);
  float* cat5 = alloc((long)RT * N2 * 1536);
  float* c5   = alloc((long)RT * N2 * 512);
  float* cat6 = alloc((long)RT * N3 * 768);
  float* c6   = alloc((long)RT * N3 * 256);
  float* cat7 = alloc((long)RT * N4 * 384);
  float* c7   = alloc((long)RT * N4 * 128);
  float* cat8 = alloc((long)RT * N5 * 192);
  float* c8   = alloc((long)RT * N5 * 64);
  float* logits = alloc((long)RT * N5);

  // ---- encoder ----
  launch_sconv(frame, neigh5, W[0], bb[0], c0, N5, 3, 64, 1, stream);
  { long t = (long)RT * N4 * 64;
    pool_ring<<<gsz(t), 256, 0, stream>>>(c0, neigh5, p0, N5, N4, 64); }
  launch_sconv(p0, neigh4, W[1], bb[1], c1, N4, 64, 128, 1, stream);
  { long t = (long)RT * N3 * 128;
    pool_ring<<<gsz(t), 256, 0, stream>>>(c1, neigh4, p1, N4, N3, 128); }
  launch_sconv(p1, neigh3, W[2], bb[2], c2, N3, 128, 256, 1, stream);
  { long t = (long)RT * N2 * 256;
    pool_ring<<<gsz(t), 256, 0, stream>>>(c2, neigh3, p2, N3, N2, 256); }
  launch_sconv(p2, neigh2, W[3], bb[3], c3, N2, 256, 512, 1, stream);
  { long t = (long)RT * N1 * 512;
    pool_ring<<<gsz(t), 256, 0, stream>>>(c3, neigh2, p3, N2, N1, 512); }
  launch_sconv(p3, neigh1, W[4], bb[4], c4, N1, 512, 1024, 1, stream);

  // ---- decoder ----
  { long t = (long)RT * N2 * 1536;
    unpool_concat<<<gsz(t), 256, 0, stream>>>(c4, up2, c3, cat5, N1, N2, 1024, 512); }
  launch_sconv(cat5, neigh2, W[5], bb[5], c5, N2, 1536, 512, 1, stream);
  { long t = (long)RT * N3 * 768;
    unpool_concat<<<gsz(t), 256, 0, stream>>>(c5, up3, c2, cat6, N2, N3, 512, 256); }
  launch_sconv(cat6, neigh3, W[6], bb[6], c6, N3, 768, 256, 1, stream);
  { long t = (long)RT * N4 * 384;
    unpool_concat<<<gsz(t), 256, 0, stream>>>(c6, up4, c1, cat7, N3, N4, 256, 128); }
  launch_sconv(cat7, neigh4, W[7], bb[7], c7, N4, 384, 128, 1, stream);
  { long t = (long)RT * N5 * 192;
    unpool_concat<<<gsz(t), 256, 0, stream>>>(c7, up5, c0, cat8, N4, N5, 128, 64); }
  launch_sconv(cat8, neigh5, W[8], bb[8], c8, N5, 192, 64, 1, stream);

  // ---- head + softmax ----
  conv_out1<<<(RT * N5 + 255) / 256, 256, 0, stream>>>(c8, neigh5, W[9], bb[9], logits, N5);
  softmax_rows<<<RT, 256, 0, stream>>>(logits, (float*)d_out, N5);
}
